// MultiHeadAttention_46093589021200
// MI455X (gfx1250) — matrix operand order repaired and checked by reading, not compiled, not run on hardware
//
#include <hip/hip_runtime.h>

// ---------------------------------------------------------------------------
// MHA forward for MI455X (gfx1250, wave32, WMMA + async global->LDS copies).
// B=4, S=2048, E=1024, H=16, D=64.
// ---------------------------------------------------------------------------

typedef __attribute__((ext_vector_type(16))) __bf16 v16bf;
typedef __attribute__((ext_vector_type(8)))  float  v8f;

#define E_DIM 1024
#define N3    3072
#define S_LEN 2048
#define HEADS 16
#define HD    64
#define BATCH 4

static __device__ __forceinline__ unsigned short f2bf(float f) {
  unsigned u = __builtin_bit_cast(unsigned, f);
  u += 0x7FFFu + ((u >> 16) & 1u);          // round-to-nearest-even
  return (unsigned short)(u >> 16);
}

union Frag {
  unsigned u[8];
  v16bf    v;
};

static __device__ __forceinline__ v8f wmma_bf16(v16bf a, v16bf b, v8f c) {
  return __builtin_amdgcn_wmma_f32_16x16x32_bf16(false, a, false, b, (short)0, c,
                                                 false, false);
}

static __device__ __forceinline__ v8f vzero8() {
  v8f z = {0.f, 0.f, 0.f, 0.f, 0.f, 0.f, 0.f, 0.f};
  return z;
}

// Async global->LDS 16-byte copy (per lane), tracked by ASYNCcnt.
// lds_off: LDS byte offset (low 32 bits of generic shared pointer).
static __device__ __forceinline__ void async_b128(unsigned lds_off,
                                                  const void* gptr) {
  asm volatile("global_load_async_to_lds_b128 %0, %1, off"
               :: "v"(lds_off), "v"(gptr)
               : "memory");
}

static __device__ __forceinline__ void wait_async0() {
#if __has_builtin(__builtin_amdgcn_s_wait_asynccnt)
  __builtin_amdgcn_s_wait_asynccnt(0);
#else
  asm volatile("s_wait_asynccnt 0x0" ::: "memory");
#endif
}

static __device__ __forceinline__ unsigned lds_off_of(const void* p) {
  return (unsigned)(size_t)p;   // LDS generic addr: offset lives in [31:0]
}

// ---------------------------------------------------------------------------
// Prepass A: elementwise f32 -> bf16 (x).
// ---------------------------------------------------------------------------
__global__ __launch_bounds__(256) void cvt_bf16_kernel(
    const float* __restrict__ in, unsigned short* __restrict__ out) {
  const int i = (blockIdx.x * 256 + threadIdx.x) * 4;
  const float4 v = *(const float4*)(in + i);
  out[i + 0] = f2bf(v.x);
  out[i + 1] = f2bf(v.y);
  out[i + 2] = f2bf(v.z);
  out[i + 3] = f2bf(v.w);
}

// ---------------------------------------------------------------------------
// Prepass B: W [K][N] f32 -> Wt [N][K] bf16 (transpose through LDS).
// ---------------------------------------------------------------------------
__global__ __launch_bounds__(256) void cvtT_kernel(
    const float* __restrict__ W, unsigned short* __restrict__ Wt,
    int K, int N) {
  __shared__ unsigned short tile[64][65];
  const int tid = threadIdx.x;
  const int kT = blockIdx.x * 64, nT = blockIdx.y * 64;
#pragma unroll
  for (int t = 0; t < 16; ++t) {
    int idx = tid + t * 256;
    int r = idx >> 6, c = idx & 63;           // r: k, c: n  (coalesced on n)
    tile[r][c] = f2bf(W[(size_t)(kT + r) * N + nT + c]);
  }
  __syncthreads();
#pragma unroll
  for (int t = 0; t < 16; ++t) {
    int idx = tid + t * 256;
    int r = idx >> 6, c = idx & 63;           // r: n, c: k  (coalesced on k)
    Wt[(size_t)(nT + r) * K + kT + c] = tile[c][r];
  }
}

// ---------------------------------------------------------------------------
// Kernel 1: QKV projection, all-bf16, async double-buffered.
// A = Xb [8192][1024] bf16, B = Wqt [3072][1024] bf16 (N-major).
// 256 threads (8 waves, 4x2), tile 128x64, k-slab 32.
// LDS stride 72 bf16 (144B = 36 dwords) -> conflict-free b128 frags.
// ---------------------------------------------------------------------------
__global__ __launch_bounds__(256) void qkv_gemm_kernel(
    const unsigned short* __restrict__ Xb, const unsigned short* __restrict__ Wt,
    const float* __restrict__ bias, unsigned short* __restrict__ Qb,
    unsigned short* __restrict__ Kb, unsigned short* __restrict__ Vb) {
  __shared__ unsigned short sA[2][128 * 72];
  __shared__ unsigned short sB[2][64 * 72];

  const int tid  = threadIdx.x;
  const int wave = tid >> 5, lane = tid & 31;
  const int hm = lane & 15, hk = lane >> 4;
  const int hi8 = hk * 8;
  const int mT = blockIdx.x * 128, nT = blockIdx.y * 64;
  const int mB = (wave >> 1) * 32, nB = (wave & 1) * 32;

  auto stage = [&](int kT, int buf) {
#pragma unroll
    for (int i = 0; i < 2; ++i) {                    // A: 128 rows x 4 segs
      int c = tid + i * 256;
      int row = c >> 2, seg = c & 3;
      async_b128(lds_off_of(&sA[buf][row * 72 + seg * 8]),
                 Xb + (size_t)(mT + row) * E_DIM + kT + seg * 8);
    }
    {                                                // B: 64 rows x 4 segs
      int row = tid >> 2, seg = tid & 3;
      async_b128(lds_off_of(&sB[buf][row * 72 + seg * 8]),
                 Wt + (size_t)(nT + row) * E_DIM + kT + seg * 8);
    }
  };

  v8f acc[2][2];
  acc[0][0] = vzero8(); acc[0][1] = vzero8();
  acc[1][0] = vzero8(); acc[1][1] = vzero8();

  stage(0, 0);
  wait_async0();
  __syncthreads();

  for (int kT = 0, it = 0; kT < E_DIM; kT += 32, ++it) {
    const int cur = it & 1;
    if (kT + 32 < E_DIM) stage(kT + 32, cur ^ 1);

    const unsigned* a32 = (const unsigned*)sA[cur];
    const unsigned* b32 = (const unsigned*)sB[cur];
    Frag a[2], b[2];
#pragma unroll
    for (int i2 = 0; i2 < 2; ++i2) {
      int base = (mB + i2 * 16 + hm) * 36 + hk * 8;
#pragma unroll
      for (int i = 0; i < 4; ++i) {
        a[i2].u[i]     = a32[base + i];
        a[i2].u[4 + i] = a32[base + 4 + i];
      }
    }
#pragma unroll
    for (int j2 = 0; j2 < 2; ++j2) {
      int base = (nB + j2 * 16 + hm) * 36 + hk * 8;
#pragma unroll
      for (int i = 0; i < 8; ++i) b[j2].u[i] = b32[base + i];
    }
#pragma unroll
    for (int i2 = 0; i2 < 2; ++i2)
#pragma unroll
      for (int j2 = 0; j2 < 2; ++j2)
        acc[i2][j2] = wmma_bf16(a[i2].v, b[j2].v, acc[i2][j2]);

    wait_async0();
    __syncthreads();
  }

  // Scatter to Q/K/V [B,H,S,D]; 64-wide N tile lies in one (head, q/k/v) slice.
  const int h   = nT / 192;
  const int sel = (nT % 192) / 64;
  unsigned short* dst = (sel == 0) ? Qb : ((sel == 1) ? Kb : Vb);
#pragma unroll
  for (int i2 = 0; i2 < 2; ++i2) {
#pragma unroll
    for (int j2 = 0; j2 < 2; ++j2) {
      const int d  = nB + j2 * 16 + hm;
      const float bv = bias[nT + d];
#pragma unroll
      for (int r = 0; r < 8; ++r) {
        int m  = mT + mB + i2 * 16 + r + hi8;
        int b_ = m >> 11, s_ = m & (S_LEN - 1);
        dst[(((size_t)b_ * HEADS + h) * S_LEN + s_) * HD + d] =
            f2bf(acc[i2][j2][r] + bv);
      }
    }
  }
}

// ---------------------------------------------------------------------------
// Kernel 2: fused causal flash attention per (b,h, 64-query tile).
// ---------------------------------------------------------------------------
__global__ __launch_bounds__(128) void attn_kernel(
    const unsigned short* __restrict__ Qb, const unsigned short* __restrict__ Kb,
    const unsigned short* __restrict__ Vb, unsigned short* __restrict__ Ob) {
  __shared__ unsigned short sV[64 * 68];        // V transposed [d][key]
  __shared__ unsigned short sP[4][16 * 68];     // per-wave P tile [16][64]

  const int tid  = threadIdx.x;
  const int wave = tid >> 5, lane = tid & 31;
  const int hm = lane & 15, hk = lane >> 4;
  const int hi8 = hk * 8;
  const int bh = blockIdx.y;
  const int qT = blockIdx.x * 64;
  const int qRow = qT + wave * 16;

  const unsigned* qp =
      (const unsigned*)(Qb + ((size_t)bh * S_LEN + qRow + hm) * HD);
  Frag qf[2];
#pragma unroll
  for (int c2 = 0; c2 < 2; ++c2) {
    int base = c2 * 16 + hk * 8;
#pragma unroll
    for (int i = 0; i < 4; ++i) {
      qf[c2].u[i]     = qp[base + i];
      qf[c2].u[4 + i] = qp[base + 4 + i];
    }
  }

  float mrow[8], lrow[8];
#pragma unroll
  for (int r = 0; r < 8; ++r) { mrow[r] = -3.0e38f; lrow[r] = 0.f; }
  v8f o[4];
  o[0] = vzero8(); o[1] = vzero8(); o[2] = vzero8(); o[3] = vzero8();

  const unsigned* sV32 = (const unsigned*)sV;
  const unsigned* sPw  = (const unsigned*)sP[wave];

  const int jMax = qT >> 6;
  for (int j = 0; j <= jMax; ++j) {
    const int kBase = j * 64;
    __syncthreads();
#pragma unroll
    for (int t = 0; t < 32; ++t) {              // stage V transposed in LDS
      int idx = tid + t * 128;
      int key = idx >> 6, d = idx & 63;
      sV[d * 68 + key] = Vb[((size_t)bh * S_LEN + kBase + key) * HD + d];
    }
    __syncthreads();

    v8f s[4];
    s[0] = vzero8(); s[1] = vzero8(); s[2] = vzero8(); s[3] = vzero8();
#pragma unroll
    for (int n = 0; n < 4; ++n) {
      const unsigned* kp = (const unsigned*)(
          Kb + ((size_t)bh * S_LEN + kBase + n * 16 + hm) * HD);
#pragma unroll
      for (int c2 = 0; c2 < 2; ++c2) {
        Frag kf;
        int base = c2 * 16 + hk * 8;
#pragma unroll
        for (int i = 0; i < 8; ++i) kf.u[i] = kp[base + i];
        s[n] = wmma_bf16(qf[c2].v, kf.v, s[n]);
      }
    }

    const bool diag = (j == jMax);
#pragma unroll
    for (int n = 0; n < 4; ++n)
#pragma unroll
      for (int r = 0; r < 8; ++r) {
        float val = s[n][r] * 0.125f;           // 1/sqrt(64)
        if (diag && (kBase + n * 16 + hm > qRow + r + hi8)) val = -3.0e38f;
        s[n][r] = val;
      }

#pragma unroll
    for (int r = 0; r < 8; ++r) {
      float mx = fmaxf(fmaxf(s[0][r], s[1][r]), fmaxf(s[2][r], s[3][r]));
      mx = fmaxf(mx, __shfl_xor(mx, 1, 16));
      mx = fmaxf(mx, __shfl_xor(mx, 2, 16));
      mx = fmaxf(mx, __shfl_xor(mx, 4, 16));
      mx = fmaxf(mx, __shfl_xor(mx, 8, 16));
      float mnew = fmaxf(mrow[r], mx);
      float corr = __expf(mrow[r] - mnew);
      mrow[r] = mnew;
      float rs = 0.f;
#pragma unroll
      for (int n = 0; n < 4; ++n) {
        float p = __expf(s[n][r] - mnew);
        s[n][r] = p;
        rs += p;
      }
      rs += __shfl_xor(rs, 1, 16);
      rs += __shfl_xor(rs, 2, 16);
      rs += __shfl_xor(rs, 4, 16);
      rs += __shfl_xor(rs, 8, 16);
      lrow[r] = lrow[r] * corr + rs;
#pragma unroll
      for (int n = 0; n < 4; ++n) o[n][r] *= corr;
    }

#pragma unroll
    for (int n = 0; n < 4; ++n)
#pragma unroll
      for (int r = 0; r < 8; ++r)
        sP[wave][(r + hi8) * 68 + n * 16 + hm] = f2bf(s[n][r]);
    __syncthreads();

    Frag pf[2];
#pragma unroll
    for (int c2 = 0; c2 < 2; ++c2) {
      int base = hm * 34 + c2 * 16 + hk * 8;
#pragma unroll
      for (int i = 0; i < 4; ++i) {
        pf[c2].u[i]     = sPw[base + i];
        pf[c2].u[4 + i] = sPw[base + 4 + i];
      }
    }
#pragma unroll
    for (int n = 0; n < 4; ++n) {
      int rbase = (n * 16 + hm) * 34;
#pragma unroll
      for (int c2 = 0; c2 < 2; ++c2) {
        Frag vf;
        int base = rbase + c2 * 16 + hk * 8;
#pragma unroll
        for (int i = 0; i < 8; ++i) vf.u[i] = sV32[base + i];
        o[n] = wmma_bf16(pf[c2].v, vf.v, o[n]);
      }
    }
  }

#pragma unroll
  for (int r = 0; r < 8; ++r) {
    float inv = 1.0f / lrow[r];
    size_t row = (size_t)bh * S_LEN + qRow + r + hi8;
#pragma unroll
    for (int n = 0; n < 4; ++n)
      Ob[row * HD + n * 16 + hm] = f2bf(o[n][r] * inv);
  }
}

// ---------------------------------------------------------------------------
// Kernel 3: output projection, all-bf16, async double-buffered.
// A = attn bf16 (flat == reference's no-transpose reshape), B = Wot [N][K].
// ---------------------------------------------------------------------------
__global__ __launch_bounds__(256) void out_proj_kernel(
    const unsigned short* __restrict__ Ab, const unsigned short* __restrict__ Wt,
    const float* __restrict__ bias, float* __restrict__ out) {
  __shared__ unsigned short sA[2][128 * 72];
  __shared__ unsigned short sB[2][64 * 72];

  const int tid  = threadIdx.x;
  const int wave = tid >> 5, lane = tid & 31;
  const int hm = lane & 15, hk = lane >> 4;
  const int hi8 = hk * 8;
  const int mT = blockIdx.x * 128, nT = blockIdx.y * 64;
  const int mB = (wave >> 1) * 32, nB = (wave & 1) * 32;

  auto stage = [&](int kT, int buf) {
#pragma unroll
    for (int i = 0; i < 2; ++i) {
      int c = tid + i * 256;
      int row = c >> 2, seg = c & 3;
      async_b128(lds_off_of(&sA[buf][row * 72 + seg * 8]),
                 Ab + (size_t)(mT + row) * E_DIM + kT + seg * 8);
    }
    {
      int row = tid >> 2, seg = tid & 3;
      async_b128(lds_off_of(&sB[buf][row * 72 + seg * 8]),
                 Wt + (size_t)(nT + row) * E_DIM + kT + seg * 8);
    }
  };

  v8f acc[2][2];
  acc[0][0] = vzero8(); acc[0][1] = vzero8();
  acc[1][0] = vzero8(); acc[1][1] = vzero8();

  stage(0, 0);
  wait_async0();
  __syncthreads();

  for (int kT = 0, it = 0; kT < E_DIM; kT += 32, ++it) {
    const int cur = it & 1;
    if (kT + 32 < E_DIM) stage(kT + 32, cur ^ 1);

    const unsigned* a32 = (const unsigned*)sA[cur];
    const unsigned* b32 = (const unsigned*)sB[cur];
    Frag a[2], b[2];
#pragma unroll
    for (int i2 = 0; i2 < 2; ++i2) {
      int base = (mB + i2 * 16 + hm) * 36 + hk * 8;
#pragma unroll
      for (int i = 0; i < 4; ++i) {
        a[i2].u[i]     = a32[base + i];
        a[i2].u[4 + i] = a32[base + 4 + i];
      }
    }
#pragma unroll
    for (int j2 = 0; j2 < 2; ++j2) {
      int base = (nB + j2 * 16 + hm) * 36 + hk * 8;
#pragma unroll
      for (int i = 0; i < 8; ++i) b[j2].u[i] = b32[base + i];
    }
#pragma unroll
    for (int i2 = 0; i2 < 2; ++i2)
#pragma unroll
      for (int j2 = 0; j2 < 2; ++j2)
        acc[i2][j2] = wmma_bf16(a[i2].v, b[j2].v, acc[i2][j2]);

    wait_async0();
    __syncthreads();
  }

#pragma unroll
  for (int i2 = 0; i2 < 2; ++i2)
#pragma unroll
    for (int j2 = 0; j2 < 2; ++j2) {
      const int nG = nT + nB + j2 * 16 + hm;
      const float bv = bias[nG];
#pragma unroll
      for (int r = 0; r < 8; ++r) {
        int m = mT + mB + i2 * 16 + r + hi8;
        out[(size_t)m * E_DIM + nG] = acc[i2][j2][r] + bv;
      }
    }
}

// ---------------------------------------------------------------------------
extern "C" void kernel_launch(void* const* d_in, const int* in_sizes, int n_in,
                              void* d_out, int out_size, void* d_ws,
                              size_t ws_size, hipStream_t stream) {
  const float* x    = (const float*)d_in[0];
  // d_in[1]: causal mask, handled analytically in-kernel.
  const float* Wqkv = (const float*)d_in[2];
  const float* bqkv = (const float*)d_in[3];
  const float* Wo   = (const float*)d_in[4];
  const float* bo   = (const float*)d_in[5];
  float* out = (float*)d_out;

  const size_t nX    = (size_t)BATCH * S_LEN * E_DIM;        // 8,388,608
  const size_t nHead = (size_t)BATCH * HEADS * S_LEN * HD;   // 8,388,608

  unsigned short* Xb  = (unsigned short*)d_ws;
  unsigned short* Wqt = Xb + nX;                  // [3072][1024]
  unsigned short* Wot = Wqt + (size_t)N3 * E_DIM; // [1024][1024]
  unsigned short* Qb  = Wot + (size_t)E_DIM * E_DIM;
  unsigned short* Kb  = Qb + nHead;
  unsigned short* Vb  = Kb + nHead;
  unsigned short* Ob  = Vb + nHead;

  // Prepass: bf16 conversions (one-time, ~50 MB of traffic).
  cvt_bf16_kernel<<<dim3((unsigned)(nX / 1024)), dim3(256), 0, stream>>>(x, Xb);
  cvtT_kernel<<<dim3(E_DIM / 64, N3 / 64), dim3(256), 0, stream>>>(Wqkv, Wqt,
                                                                   E_DIM, N3);
  cvtT_kernel<<<dim3(E_DIM / 64, E_DIM / 64), dim3(256), 0, stream>>>(Wo, Wot,
                                                                      E_DIM,
                                                                      E_DIM);
  // 1) QKV projection: M=8192, N=3072 -> 64 x 48 tiles of 128x64
  qkv_gemm_kernel<<<dim3(64, 48), dim3(256), 0, stream>>>(Xb, Wqt, bqkv, Qb, Kb,
                                                          Vb);
  // 2) fused causal attention: 32 q-tiles x 64 (b,h) pairs
  attn_kernel<<<dim3(32, BATCH * HEADS), dim3(128), 0, stream>>>(Qb, Kb, Vb, Ob);
  // 3) output projection: M=8192, N=1024 -> 64 x 16 tiles
  out_proj_kernel<<<dim3(64, 16), dim3(256), 0, stream>>>(Ob, Wot, bo, out);
}
